// VSSEncoderLayer_3238405341500
// MI455X (gfx1250) — compile-verified, matrix operand order checked by reading, not run on hardware
//
#include <hip/hip_runtime.h>
#include <hip/hip_bf16.h>
#include <math.h>

// ---------------------------------------------------------------- constants
#define BB   4
#define HH   32
#define WW   32
#define CC   256
#define DN   512
#define NS   64          // d_state
#define RR   16          // dt_rank
#define LL   (HH*WW)     // 1024
#define MT   (BB*LL)     // 4096 rows
#define PROJ_N   (RR + 2*NS)   // 144
#define PROJ_PAD 160           // padded to 32-col multiple

typedef __attribute__((ext_vector_type(16))) __bf16 v16bf;
typedef __attribute__((ext_vector_type(8)))  __bf16 v8bf;
typedef __attribute__((ext_vector_type(8)))  float  v8f;

// ---------------------------------------------------------------- weight prep:
// fp32 [K,Ncols] -> bf16 transposed+padded [Npad,K] (rows >= Ncols zero-filled)
__global__ void wt_prep_kernel(const float* __restrict__ in, __bf16* __restrict__ out,
                               int K, int Ncols, int Npad) {
    int idx = blockIdx.x * blockDim.x + threadIdx.x;   // over Npad*K
    if (idx >= Npad * K) return;
    int k = idx % K;
    int n = idx / K;
    out[idx] = (n < Ncols) ? (__bf16)in[(size_t)k * Ncols + n] : (__bf16)0.0f;
}

// ---------------------------------------------------------------- LayerNorm over C=256 -> bf16
__global__ __launch_bounds__(256) void ln1_kernel(const float* __restrict__ src,
                                                  const float* __restrict__ g,
                                                  const float* __restrict__ b,
                                                  __bf16* __restrict__ out) {
    int row = blockIdx.x;       // 0..4095
    int t   = threadIdx.x;      // 0..255 == channel
    float v = src[(size_t)row * CC + t];
    __shared__ float red[256];
    red[t] = v; __syncthreads();
    #pragma unroll
    for (int s = 128; s > 0; s >>= 1) { if (t < s) red[t] += red[t + s]; __syncthreads(); }
    float mu = red[0] * (1.0f / CC);
    __syncthreads();
    float d = v - mu;
    red[t] = d * d; __syncthreads();
    #pragma unroll
    for (int s = 128; s > 0; s >>= 1) { if (t < s) red[t] += red[t + s]; __syncthreads(); }
    float var = red[0] * (1.0f / CC);
    out[(size_t)row * CC + t] = (__bf16)(d * rsqrtf(var + 1e-5f) * g[t] + b[t]);
}

// ---------------------------------------------------------------- bf16 WMMA GEMM, direct-fragment
// out[M,Ncols](f32,ldOut) = A[M,KD](bf16 row-major) @ W^T[Npad,KD](bf16) (+bias)(+residual)
// Fragment layout (ISA 7.12.2, 16-bit): element e -> K = ((e&8)<<1)+(half<<3)+(e&7),
// i.e. two contiguous 8-element runs -> two b128 loads per fragment. No LDS needed.
template<int KD>
__global__ __launch_bounds__(256) void gemm_bf16_wmma(
    const __bf16* __restrict__ A, const __bf16* __restrict__ WT,
    const float* __restrict__ bias, const float* __restrict__ residual,
    float* __restrict__ out, int Ncols, int ldOut)
{
    const int wave = threadIdx.x >> 5, lane = threadIdx.x & 31;
    const int wm = wave & 3;                 // 4 row tiles  (64 rows / block)
    const int wn = wave >> 2;                // 2 col tiles  (32 cols / block)
    const int lr = lane & 15, half = lane >> 4;
    const int row = blockIdx.x * 64 + wm * 16 + lr;
    const int col = blockIdx.y * 32 + wn * 16 + lr;   // < Npad by construction

    const v8bf* ap = (const v8bf*)(A  + (size_t)row * KD + half * 8);
    const v8bf* bp = (const v8bf*)(WT + (size_t)col * KD + half * 8);

    v8f acc = {};
    #pragma unroll
    for (int k = 0; k < KD / 32; ++k) {
        union { v16bf v; v8bf h[2]; } a, b;
        a.h[0] = ap[4 * k];     a.h[1] = ap[4 * k + 2];
        b.h[0] = bp[4 * k];     b.h[1] = bp[4 * k + 2];
        acc = __builtin_amdgcn_wmma_f32_16x16x32_bf16(
                  false, a.v, false, b.v, (short)0, acc, false, false);
    }
    if (col < Ncols) {
        float bv = bias ? bias[col] : 0.0f;
        #pragma unroll
        for (int j = 0; j < 8; ++j) {
            int r = blockIdx.x * 64 + wm * 16 + half * 8 + j;   // D: row = j + 8*half
            size_t idx = (size_t)r * ldOut + col;
            float v = acc[j] + bv;
            if (residual) v += residual[idx];
            out[idx] = v;
        }
    }
}

// ---------------------------------------------------------------- depthwise 3x3 conv + GELU(tanh)
__global__ __launch_bounds__(256) void conv_gelu_kernel(const float* __restrict__ x,
                                                        const float* __restrict__ cw,
                                                        const float* __restrict__ cb,
                                                        float* __restrict__ xg,
                                                        __bf16* __restrict__ xgb) {
    int idx = blockIdx.x * blockDim.x + threadIdx.x;     // B*H*W*Dn
    if (idx >= BB * HH * WW * DN) return;
    int d = idx & (DN - 1);
    int w = (idx >> 9)  & (WW - 1);
    int h = (idx >> 14) & (HH - 1);
    int b = idx >> 19;
    float s = 0.f;
    #pragma unroll
    for (int dy = -1; dy <= 1; ++dy) {
        int hh = h + dy; if (hh < 0 || hh >= HH) continue;
        #pragma unroll
        for (int dx = -1; dx <= 1; ++dx) {
            int ww = w + dx; if (ww < 0 || ww >= WW) continue;
            s += x[((size_t)((b * HH + hh) * WW + ww)) * DN + d]
               * cw[((dy + 1) * 3 + (dx + 1)) * DN + d];
        }
    }
    s += cb[d];
    float u = s * (s * s * 0.044715f + 1.0f) * 0.7978845608028654f;
    float g = 0.5f * s * (1.0f + tanhf(u));
    xg[idx]  = g;
    xgb[idx] = (__bf16)g;
}

// ---------------------------------------------------------------- dt = softplus(dt_raw @ w_dt + bias)
__global__ __launch_bounds__(256) void dt_kernel(const float* __restrict__ proj,
                                                 const float* __restrict__ w_dt,
                                                 const float* __restrict__ dt_bias,
                                                 float* __restrict__ dt) {
    int idx = blockIdx.x * blockDim.x + threadIdx.x;     // MT*DN
    if (idx >= MT * DN) return;
    int d = idx & (DN - 1);
    int m = idx >> 9;
    float s = dt_bias[d];
    #pragma unroll
    for (int r = 0; r < RR; ++r)
        s += proj[(size_t)m * PROJ_PAD + r] * w_dt[r * DN + d];
    dt[idx] = (s > 20.f) ? s : log1pf(__expf(s));
}

// ---------------------------------------------------------------- selective scan
// one wave32 per (b,d); lane holds states n=lane and n=lane+32.
// 8 waves/block share per-(b,l) B/C vectors + per-d x,dt scalars via double-buffered LDS:
// one barrier per timestep. grid = B * (DN/8) = 256 blocks.
__global__ __launch_bounds__(256) void scan_kernel(const float* __restrict__ proj,
                                                   const float* __restrict__ xg,
                                                   const float* __restrict__ dt,
                                                   const float* __restrict__ A_log,
                                                   float* __restrict__ ys) {
    const int blk  = blockIdx.x;
    const int b    = blk >> 6;
    const int dgrp = blk & 63;
    const int tid  = threadIdx.x;
    const int wave = tid >> 5, lane = tid & 31;
    const int d    = dgrp * 8 + wave;

    const float A0 = -__expf(A_log[d * NS + lane]);
    const float A1 = -__expf(A_log[d * NS + 32 + lane]);
    float h0 = 0.f, h1 = 0.f;

    __shared__ float bc[2][144];        // [0..127]=B,C  [128..135]=x  [136..143]=dt
    const float* projB = proj + (size_t)b * LL * PROJ_PAD + RR;
    const float* xgb   = xg + (size_t)b * LL * DN + dgrp * 8;
    const float* dtb   = dt + (size_t)b * LL * DN + dgrp * 8;
    float*       ysb   = ys + (size_t)b * LL * DN + d;

    // preload l = 0
    if (tid < 128)       bc[0][tid]       = projB[tid];
    else if (tid < 136)  bc[0][tid]       = xgb[tid - 128];
    else if (tid < 144)  bc[0][tid]       = dtb[tid - 136];
    __syncthreads();

    for (int l = 0; l < LL; ++l) {
        const int cur = l & 1, nxt = cur ^ 1;
        float xt  = bc[cur][128 + wave];
        float dtt = bc[cur][136 + wave];
        float B0 = bc[cur][lane],      B1 = bc[cur][32 + lane];
        float C0 = bc[cur][NS + lane], C1 = bc[cur][NS + 32 + lane];

        // stage l+1 into the other buffer (disjoint from current reads)
        if (l + 1 < LL) {
            size_t lo = (size_t)(l + 1);
            if (tid < 128)      bc[nxt][tid] = projB[lo * PROJ_PAD + tid];
            else if (tid < 136) bc[nxt][tid] = xgb[lo * DN + (tid - 128)];
            else if (tid < 144) bc[nxt][tid] = dtb[lo * DN + (tid - 136)];
        }
        if (tid == 0 && l + 8 < LL)
            __builtin_prefetch(&projB[(size_t)(l + 8) * PROJ_PAD], 0, 1);

        float dx = dtt * xt;
        h0 = __expf(dtt * A0) * h0 + dx * B0;
        h1 = __expf(dtt * A1) * h1 + dx * B1;
        float y = h0 * C0 + h1 * C1;
        #pragma unroll
        for (int off = 16; off > 0; off >>= 1) y += __shfl_xor(y, off, 32);
        if (lane == 0) ysb[(size_t)l * DN] = y;
        __syncthreads();
    }
}

// ---------------------------------------------------------------- y = ys + xg*D; LN over Dn -> bf16
__global__ __launch_bounds__(512) void ln2_kernel(const float* __restrict__ ys,
                                                  const float* __restrict__ xg,
                                                  const float* __restrict__ Dsk,
                                                  const float* __restrict__ g,
                                                  const float* __restrict__ b,
                                                  __bf16* __restrict__ out) {
    int row = blockIdx.x;
    int t   = threadIdx.x;      // 0..511
    float v = ys[(size_t)row * DN + t] + xg[(size_t)row * DN + t] * Dsk[t];
    __shared__ float red[512];
    red[t] = v; __syncthreads();
    #pragma unroll
    for (int s = 256; s > 0; s >>= 1) { if (t < s) red[t] += red[t + s]; __syncthreads(); }
    float mu = red[0] * (1.0f / DN);
    __syncthreads();
    float d = v - mu;
    red[t] = d * d; __syncthreads();
    #pragma unroll
    for (int s = 256; s > 0; s >>= 1) { if (t < s) red[t] += red[t + s]; __syncthreads(); }
    float var = red[0] * (1.0f / DN);
    out[(size_t)row * DN + t] = (__bf16)(d * rsqrtf(var + 1e-5f) * g[t] + b[t]);
}

// ---------------------------------------------------------------- host
static inline size_t alignup(size_t x) { return (x + 255) & ~(size_t)255; }

extern "C" void kernel_launch(void* const* d_in, const int* in_sizes, int n_in,
                              void* d_out, int out_size, void* d_ws, size_t ws_size,
                              hipStream_t stream) {
    const float* src     = (const float*)d_in[0];
    const float* ln1_g   = (const float*)d_in[1];
    const float* ln1_b   = (const float*)d_in[2];
    const float* w_in    = (const float*)d_in[3];
    const float* b_in    = (const float*)d_in[4];
    const float* conv_w  = (const float*)d_in[5];
    const float* conv_b  = (const float*)d_in[6];
    const float* w_x     = (const float*)d_in[7];
    const float* w_dt    = (const float*)d_in[8];
    const float* dt_bias = (const float*)d_in[9];
    const float* A_log   = (const float*)d_in[10];
    const float* D_skip  = (const float*)d_in[11];
    const float* ln2_g   = (const float*)d_in[12];
    const float* ln2_b   = (const float*)d_in[13];
    const float* w_out   = (const float*)d_in[14];
    const float* b_out   = (const float*)d_in[15];
    float* out = (float*)d_out;

    // workspace carve-up
    char* p = (char*)d_ws;
    __bf16* xln_bf   = (__bf16*)p; p += alignup((size_t)MT * CC * 2);
    __bf16* w_in_t   = (__bf16*)p; p += alignup((size_t)DN * CC * 2);        // [512][256]
    __bf16* w_x_t    = (__bf16*)p; p += alignup((size_t)PROJ_PAD * DN * 2);  // [160][512]
    __bf16* w_out_t  = (__bf16*)p; p += alignup((size_t)CC * DN * 2);        // [256][512]
    float*  x1       = (float*)p;  p += alignup((size_t)MT * DN * 4);
    float*  xg       = (float*)p;  p += alignup((size_t)MT * DN * 4);
    __bf16* xg_bf    = (__bf16*)p; p += alignup((size_t)MT * DN * 2);
    float*  proj     = (float*)p;  p += alignup((size_t)MT * PROJ_PAD * 4);
    float*  dtv      = (float*)p;  p += alignup((size_t)MT * DN * 4);
    float*  ysv      = (float*)p;  p += alignup((size_t)MT * DN * 4);
    __bf16* yln_bf   = (__bf16*)p; p += alignup((size_t)MT * DN * 2);
    (void)ws_size; (void)n_in; (void)in_sizes; (void)out_size;

    // 0) weight prep: fp32 [K,N] -> bf16 transposed+padded [Npad,K]
    wt_prep_kernel<<<(DN * CC + 255) / 256, 256, 0, stream>>>(w_in, w_in_t, CC, DN, DN);
    wt_prep_kernel<<<(PROJ_PAD * DN + 255) / 256, 256, 0, stream>>>(w_x, w_x_t, DN, PROJ_N, PROJ_PAD);
    wt_prep_kernel<<<(CC * DN + 255) / 256, 256, 0, stream>>>(w_out, w_out_t, DN, CC, CC);

    // 1) LayerNorm1 -> bf16
    ln1_kernel<<<MT, 256, 0, stream>>>(src, ln1_g, ln1_b, xln_bf);

    // 2) x1 = xln @ w_in + b_in    [4096x256 @ 256x512]
    gemm_bf16_wmma<CC><<<dim3(MT / 64, DN / 32), 256, 0, stream>>>(
        xln_bf, w_in_t, b_in, nullptr, x1, DN, DN);

    // 3) depthwise conv 3x3 + GELU
    conv_gelu_kernel<<<(MT * DN + 255) / 256, 256, 0, stream>>>(x1, conv_w, conv_b, xg, xg_bf);

    // 4) proj = xg @ w_x           [4096x512 @ 512x144], padded to 160 cols
    gemm_bf16_wmma<DN><<<dim3(MT / 64, PROJ_PAD / 32), 256, 0, stream>>>(
        xg_bf, w_x_t, nullptr, nullptr, proj, PROJ_N, PROJ_PAD);

    // 5) dt = softplus(dt_raw @ w_dt + dt_bias)
    dt_kernel<<<(MT * DN + 255) / 256, 256, 0, stream>>>(proj, w_dt, dt_bias, dtv);

    // 6) selective scan: 2048 independent (b,d) wave32 chains
    scan_kernel<<<BB * (DN / 8), 256, 0, stream>>>(proj, xg, dtv, A_log, ysv);

    // 7) skip + LayerNorm2 -> bf16
    ln2_kernel<<<MT, 512, 0, stream>>>(ysv, xg, D_skip, ln2_g, ln2_b, yln_bf);

    // 8) out = src + yln @ w_out + b_out   [4096x512 @ 512x256], fused residual
    gemm_bf16_wmma<DN><<<dim3(MT / 64, CC / 32), 256, 0, stream>>>(
        yln_bf, w_out_t, b_out, src, out, CC, CC);
}
